// PrototypicalContrastiveLoss_72808285602381
// MI455X (gfx1250) — matrix operand order repaired and checked by reading, not compiled, not run on hardware
//
#include <hip/hip_runtime.h>
#include <hip/hip_bf16.h>

typedef __attribute__((ext_vector_type(16))) __bf16 v16bf;
typedef __attribute__((ext_vector_type(8)))  float  v8f;

#define NORM_EPS 1e-12f
#define INV_T    (1.0f / 0.3f)
#define MOM      0.5f

union BF8  { __bf16 h[8];  uint4 q;    };
union Frag { v16bf  v;     uint4 q[2]; };

__device__ __forceinline__ float waveRedSum32(float x) {
#pragma unroll
  for (int m = 16; m >= 1; m >>= 1) x += __shfl_xor(x, m, 32);
  return x;
}

// ---------------------------------------------------------------------------
// Pass 1: row-normalize features (fp32 -> bf16 workspace) + scatter-add class
// sums/counts via global fp32 atomics. One wave32 per 256-wide row.
// ---------------------------------------------------------------------------
__global__ void norm_accum_kernel(const float* __restrict__ src,
                                  const int*   __restrict__ labels,
                                  __bf16*      __restrict__ dst,
                                  float*       __restrict__ sums,
                                  float*       __restrict__ counts,
                                  int n, int c) {
  int row  = (blockIdx.x * blockDim.x + threadIdx.x) >> 5;
  int lane = threadIdx.x & 31;
  if (row >= n) return;
  const float4* p = (const float4*)(src + (size_t)row * 256 + lane * 8);
  float4 a = p[0], b = p[1];
  float f[8] = {a.x, a.y, a.z, a.w, b.x, b.y, b.z, b.w};
  float ss = 0.f;
#pragma unroll
  for (int j = 0; j < 8; ++j) ss = fmaf(f[j], f[j], ss);
  ss = waveRedSum32(ss);
  float inv = 1.0f / fmaxf(sqrtf(ss), NORM_EPS);
  BF8 o;
#pragma unroll
  for (int j = 0; j < 8; ++j) { f[j] *= inv; o.h[j] = (__bf16)f[j]; }
  *(uint4*)(dst + (size_t)row * 256 + lane * 8) = o.q;

  int lab = labels[row];
  if (lab >= 0 && lab < c) {
    float* s = sums + (size_t)lab * 256 + lane * 8;
#pragma unroll
    for (int j = 0; j < 8; ++j) atomicAdd(s + j, f[j]);
    if (lane == 0) atomicAdd(counts + lab, 1.0f);
  }
}

// ---------------------------------------------------------------------------
// Pass 2: prototype EMA update. One wave32 per class row.
// new = normalize(MOM*p + (1-MOM)*normalize(sum/max(cnt,1))), keep p if cnt==0
// ---------------------------------------------------------------------------
__global__ void proto_update_kernel(const float* __restrict__ sums,
                                    const float* __restrict__ counts,
                                    const float* __restrict__ pin,
                                    __bf16*      __restrict__ pout,
                                    int c) {
  int cls  = (blockIdx.x * blockDim.x + threadIdx.x) >> 5;
  int lane = threadIdx.x & 31;
  if (cls >= c) return;
  float cnt  = counts[cls];
  float invc = 1.0f / fmaxf(cnt, 1.0f);
  const float* sp = sums + (size_t)cls * 256 + lane * 8;
  const float* pp = pin  + (size_t)cls * 256 + lane * 8;
  float m[8], p[8], ss = 0.f;
#pragma unroll
  for (int j = 0; j < 8; ++j) {
    m[j] = sp[j] * invc; p[j] = pp[j]; ss = fmaf(m[j], m[j], ss);
  }
  ss = waveRedSum32(ss);
  float inv = 1.0f / fmaxf(sqrtf(ss), NORM_EPS);
  float ns = 0.f;
#pragma unroll
  for (int j = 0; j < 8; ++j) {
    m[j] = fmaf(MOM, p[j], (1.0f - MOM) * (m[j] * inv));
    ns   = fmaf(m[j], m[j], ns);
  }
  ns = waveRedSum32(ns);
  float invn = 1.0f / fmaxf(sqrtf(ns), NORM_EPS);
  BF8 o;
#pragma unroll
  for (int j = 0; j < 8; ++j) {
    float v = (cnt > 0.f) ? m[j] * invn : p[j];
    o.h[j] = (__bf16)v;
  }
  *(uint4*)(pout + (size_t)cls * 256 + lane * 8) = o.q;
}

// ---------------------------------------------------------------------------
// Pass 3: WMMA GEMM + fused softmax-NLL. Prototypes staged into LDS via the
// CDNA5 async global->LDS path (ASYNCcnt, no VGPR round-trip). Each wave owns
// 16-row blocks: A (16x256 bf16) preloaded into 8 frags, column tiles of 16
// classes via v_wmma_f32_16x16x32_bf16 (K-loop unrolled 8).
// |logit| <= 1/0.3 so logsumexp needs no max-subtraction.
// ---------------------------------------------------------------------------
__global__ void proto_loss_kernel(const __bf16* __restrict__ feat,
                                  const __bf16* __restrict__ protos,
                                  const int*    __restrict__ labels,
                                  float*        __restrict__ accum,  // [0]=sum nll, [1]=count
                                  int n, int c) {
  extern __shared__ uint4 smem4[];
  __bf16* sprot = (__bf16*)smem4;
  int t = threadIdx.x;

  // Async-stage prototypes (c*512 bytes) into LDS: 16B per lane per op.
  {
    int chunks = (c * 512) / 16;
    for (int i = t; i < chunks; i += blockDim.x) {
      unsigned lds_off = (unsigned)(i * 16);
      unsigned long long ga = (unsigned long long)protos + (unsigned long long)i * 16;
      asm volatile("global_load_async_to_lds_b128 %0, %1, off"
                   :: "v"(lds_off), "v"(ga) : "memory");
    }
    asm volatile("s_wait_asynccnt 0x0" ::: "memory");
  }
  __syncthreads();

  int lane = t & 31;
  int l16  = lane & 15;
  int hi   = lane >> 4;           // 0: rows 0-7 / K-lo half, 1: rows 8-15 / K-hi half
  int gw   = blockIdx.x * (blockDim.x >> 5) + (t >> 5);
  int nw   = gridDim.x * (blockDim.x >> 5);
  int nrb  = n >> 4;

  for (int rb = gw; rb < nrb; rb += nw) {
    int r0 = rb << 4;
    const __bf16* arow = feat + (size_t)(r0 + l16) * 256;
    Frag A[8];
#pragma unroll
    for (int kk = 0; kk < 8; ++kk) {   // K blocks of 32
      int kb = kk * 32 + hi * 8;       // V0-3: K=kb..kb+7 ; V4-7: K=kb+16..kb+23
      A[kk].q[0] = *(const uint4*)(arow + kb);
      A[kk].q[1] = *(const uint4*)(arow + kb + 16);
    }
    int   lab[8];
    float lsum[8], pick[8];
#pragma unroll
    for (int r = 0; r < 8; ++r) {
      lab[r]  = labels[r0 + hi * 8 + r];  // accum VGPR r <-> row r + 8*hi
      lsum[r] = 0.f;
      pick[r] = 0.f;
    }
    for (int c0 = 0; c0 < c; c0 += 16) {
      const __bf16* brow = sprot + (size_t)(c0 + l16) * 256 + hi * 8;
      v8f acc = {0.f, 0.f, 0.f, 0.f, 0.f, 0.f, 0.f, 0.f};
#pragma unroll
      for (int kk = 0; kk < 8; ++kk) {
        Frag Bf;
        Bf.q[0] = *(const uint4*)(brow + kk * 32);      // K = kk*32+hi*8 + 0..7
        Bf.q[1] = *(const uint4*)(brow + kk * 32 + 16); //              + 16..23
        acc = __builtin_amdgcn_wmma_f32_16x16x32_bf16(
            false, A[kk].v, false, Bf.v, (short)0, acc, false, false);
      }
      int col = c0 + l16;  // this lane's column for every accum component
#pragma unroll
      for (int r = 0; r < 8; ++r) {
        float logit = acc[r] * INV_T;
        lsum[r] += __expf(logit);
        if (lab[r] == col) pick[r] += logit;
      }
    }
    // reduce across the 16 lanes sharing each row (xor masks stay in half-wave)
#pragma unroll
    for (int r = 0; r < 8; ++r) {
#pragma unroll
      for (int m = 8; m >= 1; m >>= 1) {
        lsum[r] += __shfl_xor(lsum[r], m, 32);
        pick[r] += __shfl_xor(pick[r], m, 32);
      }
    }
    if (l16 == 0) {  // lanes 0 and 16 each own 8 distinct rows
      float s = 0.f, cv = 0.f;
#pragma unroll
      for (int r = 0; r < 8; ++r) {
        if (lab[r] >= 0) { s += __logf(lsum[r]) - pick[r]; cv += 1.0f; }
      }
      atomicAdd(&accum[0], s);
      atomicAdd(&accum[1], cv);
    }
  }
}

__global__ void finalize_kernel(const float* __restrict__ acc, float* __restrict__ out) {
  out[0] = acc[0] / fmaxf(acc[1], 1.0f);
  out[1] = acc[2] / fmaxf(acc[3], 1.0f);
}

// ---------------------------------------------------------------------------
extern "C" void kernel_launch(void* const* d_in, const int* in_sizes, int n_in,
                              void* d_out, int out_size, void* d_ws, size_t ws_size,
                              hipStream_t stream) {
  const float* f_fti   = (const float*)d_in[0];
  const float* e_rcl   = (const float*)d_in[1];
  const int*   lab_fti = (const int*)d_in[2];
  const int*   lab_rcl = (const int*)d_in[3];
  const float* p_fti   = (const float*)d_in[4];
  const float* p_rcl   = (const float*)d_in[5];

  const int B     = in_sizes[0] / 256;  // 65536
  const int N     = in_sizes[1] / 256;  // 131072
  const int C_FTI = in_sizes[4] / 256;  // 128
  const int C_RCL = in_sizes[5] / 256;  // 512

  char* ws = (char*)d_ws;
  size_t off = 0;
  auto alloc = [&](size_t bytes) -> char* {
    char* p = ws + off;
    off += (bytes + 255) & ~(size_t)255;
    return p;
  };
  __bf16* fb_fti = (__bf16*)alloc((size_t)B * 256 * 2);      // 33.6 MB
  __bf16* fb_rcl = (__bf16*)alloc((size_t)N * 256 * 2);      // 67.1 MB
  __bf16* pb_fti = (__bf16*)alloc((size_t)C_FTI * 256 * 2);
  __bf16* pb_rcl = (__bf16*)alloc((size_t)C_RCL * 256 * 2);
  char* zbase = ws + off;
  float* sums_fti = (float*)alloc((size_t)C_FTI * 256 * 4);
  float* sums_rcl = (float*)alloc((size_t)C_RCL * 256 * 4);
  float* cnt_fti  = (float*)alloc((size_t)C_FTI * 4);
  float* cnt_rcl  = (float*)alloc((size_t)C_RCL * 4);
  float* accum    = (float*)alloc(4 * sizeof(float));
  size_t zbytes = (size_t)((ws + off) - zbase);
  hipMemsetAsync(zbase, 0, zbytes, stream);

  norm_accum_kernel<<<(B + 7) / 8, 256, 0, stream>>>(f_fti, lab_fti, fb_fti,
                                                     sums_fti, cnt_fti, B, C_FTI);
  norm_accum_kernel<<<(N + 7) / 8, 256, 0, stream>>>(e_rcl, lab_rcl, fb_rcl,
                                                     sums_rcl, cnt_rcl, N, C_RCL);
  proto_update_kernel<<<(C_FTI + 7) / 8, 256, 0, stream>>>(sums_fti, cnt_fti, p_fti,
                                                           pb_fti, C_FTI);
  proto_update_kernel<<<(C_RCL + 7) / 8, 256, 0, stream>>>(sums_rcl, cnt_rcl, p_rcl,
                                                           pb_rcl, C_RCL);

  // 512-thread blocks (16 waves) grid-striding over 16-row blocks: fewer
  // blocks -> 8x less LDS staging traffic, 4 waves/SIMD even at 256KB LDS.
  proto_loss_kernel<<<256, 512, C_FTI * 256 * 2, stream>>>(
      fb_fti, pb_fti, lab_fti, accum, B, C_FTI);
  proto_loss_kernel<<<256, 512, C_RCL * 256 * 2, stream>>>(
      fb_rcl, pb_rcl, lab_rcl, accum + 2, N, C_RCL);
  finalize_kernel<<<1, 1, 0, stream>>>(accum, (float*)d_out);
}
